// InfoNCE_63316407877961
// MI455X (gfx1250) — matrix operand order repaired and checked by reading, not compiled, not run on hardware
//
#include <hip/hip_runtime.h>
#include <hip/hip_bf16.h>

typedef __attribute__((ext_vector_type(16))) __bf16 v16bf;
typedef __attribute__((ext_vector_type(8)))  __bf16 v8bf;
typedef __attribute__((ext_vector_type(4)))  __bf16 v4bf;
typedef __attribute__((ext_vector_type(8)))  float  v8f;
typedef __attribute__((ext_vector_type(4)))  float  v4f;
typedef __attribute__((ext_vector_type(4)))  unsigned int v4u;
typedef __attribute__((ext_vector_type(8)))  int    v8i;
typedef __attribute__((ext_vector_type(4)))  int    v4i;

#define B_ROWS   8192      // anchors / contexts
#define K_IN     2048      // input feature dim
#define N_PROJ   512       // low-rank dim
#define TEMP_INV 10.0f     // 1/0.1

union FragBF { v16bf v; v8bf h2[2]; };

// ---------------------------------------------------------------- kernel 0
// W f32 -> bf16 (1,048,576 elements; 4 per thread, b64 stores)
__global__ void k_convert_W(const float* __restrict__ W, __bf16* __restrict__ Wbf) {
    int idx = (blockIdx.x * blockDim.x + threadIdx.x) * 4;
    v4f f = *(const v4f*)(W + idx);
    v4bf o;
    o[0] = (__bf16)f[0]; o[1] = (__bf16)f[1];
    o[2] = (__bf16)f[2]; o[3] = (__bf16)f[3];
    *(v4bf*)(Wbf + idx) = o;
}

// ---------------------------------------------------------------- kernel 1
// P[16384,512] = h[16384,2048] @ W.T  (bf16 WMMA, f32 accum, in-reg f32->bf16 for h)
__global__ void __launch_bounds__(256) k_proj(const float* __restrict__ h,
                                              const __bf16* __restrict__ Wbf,
                                              __bf16* __restrict__ Pbf) {
    int tid  = threadIdx.x;
    int wave = tid >> 5;
    int lane = tid & 31;
    int l16  = lane & 15;
    int hi   = lane >> 4;

    int tile = blockIdx.x * 8 + wave;   // 0..32767
    int mt   = tile >> 5;               // 1024 M-tiles
    int nt   = tile & 31;               // 32 N-tiles

    long arow = (long)(mt * 16 + l16);
    const float* aptr = h + arow * K_IN + hi * 8;
    long bcol = (long)(nt * 16 + l16);
    const __bf16* bptr = Wbf + bcol * K_IN + hi * 8;

    v8f c = {};
    for (int k = 0; k < K_IN; k += 32) {
        v4f f0 = *(const v4f*)(aptr + k);
        v4f f1 = *(const v4f*)(aptr + k + 4);
        v4f f2 = *(const v4f*)(aptr + k + 16);
        v4f f3 = *(const v4f*)(aptr + k + 20);
        v16bf a;
#pragma unroll
        for (int i = 0; i < 4; ++i) {
            a[i]      = (__bf16)f0[i];
            a[4 + i]  = (__bf16)f1[i];
            a[8 + i]  = (__bf16)f2[i];
            a[12 + i] = (__bf16)f3[i];
        }
        FragBF b;
        b.h2[0] = *(const v8bf*)(bptr + k);
        b.h2[1] = *(const v8bf*)(bptr + k + 16);
        c = __builtin_amdgcn_wmma_f32_16x16x32_bf16(false, a, false, b.v,
                                                    (short)0, c, false, false);
    }
#pragma unroll
    for (int r = 0; r < 8; ++r) {
        long row = (long)(mt * 16 + hi * 8 + r);
        Pbf[row * N_PROJ + nt * 16 + l16] = (__bf16)c[r];
    }
}

// ---------------------------------------------------------------- kernel 2
// norms[i] = ||P[i,:]||  (one wave per row)
__global__ void k_norms(const __bf16* __restrict__ Pbf, float* __restrict__ norms) {
    int tid  = threadIdx.x;
    int wave = tid >> 5;
    int lane = tid & 31;
    long row = (long)(blockIdx.x * 8 + wave);
    const __bf16* p = Pbf + row * N_PROJ + lane * 16;
    v8bf a = *(const v8bf*)p;
    v8bf b = *(const v8bf*)(p + 8);
    float s = 0.f;
#pragma unroll
    for (int i = 0; i < 8; ++i) {
        float x = (float)a[i]; s += x * x;
        float y = (float)b[i]; s += y * y;
    }
#pragma unroll
    for (int m = 16; m >= 1; m >>= 1) s += __shfl_xor(s, m, 32);
    if (lane == 0) norms[row] = sqrtf(s);
}

__global__ void k_zero(float* acc) { if (threadIdx.x < 8) acc[threadIdx.x] = 0.f; }

// ---------------------------------------------------------------- TDM issue
// Build a D# for a 2D bf16 tensor [8192][512], tile [16][512] (16 KB) -> LDS.
// Bitfields per CDNA5 ISA.
__device__ __forceinline__ void tdm_load_tile(unsigned lds_addr,
                                              unsigned long long gaddr) {
    v4u g0;
    g0[0] = 1u;                                             // count=1 (valid user D#)
    g0[1] = lds_addr;                                       // LDS byte address
    g0[2] = (unsigned)(gaddr & 0xFFFFFFFFu);                // global_addr[31:0]
    g0[3] = (unsigned)((gaddr >> 32) & 0x01FFFFFFu)         // global_addr[56:32]
          | (2u << 30);                                     // type=2 ("image")
    v8i g1;
    g1[0] = (int)(1u << 16);        // workgroup_mask=0, data_size=1 (2 bytes)
    g1[1] = (int)(512u << 16);      // tensor_dim0[15:0]=512 in bits[63:48]
    g1[2] = (int)(8192u << 16);     // tensor_dim0 hi=0 | tensor_dim1[15:0]=8192
    g1[3] = (int)(512u << 16);      // tensor_dim1 hi=0 | tile_dim0=512
    g1[4] = 16;                     // tile_dim1=16, tile_dim2=0
    g1[5] = 512;                    // tensor_dim0_stride = 512 elements
    g1[6] = 0;
    g1[7] = 0;
    v4i z = {0, 0, 0, 0};
#if __clang_major__ >= 23
    v8i z8 = {0, 0, 0, 0, 0, 0, 0, 0};
    __builtin_amdgcn_tensor_load_to_lds(g0, g1, z, z, z8, 0);
#else
    __builtin_amdgcn_tensor_load_to_lds(g0, g1, z, z, 0);
#endif
}

// ---------------------------------------------------------------- kernel 3
// Fused sim = scale*(wz @ wc.T) with streaming log-softmax.
// 128 blocks x 4 waves. Each wave owns a 16-row anchor block (full rows ->
// no cross-wave softmax merge). All 4 waves share wc column tiles staged in
// LDS by the Tensor Data Mover (double-buffered, prefetch 1 tile ahead):
// 4x less L2 traffic than per-wave streaming.
__global__ void __launch_bounds__(128, 1) k_sim(const __bf16* __restrict__ Pbf,
                                                const float* __restrict__ norms,
                                                float* __restrict__ acc) {
    __shared__ __bf16 bbuf[2][16][N_PROJ];   // 2 x 16 KB double buffer

    int tid  = threadIdx.x;
    int wave = tid >> 5;
    int lane = tid & 31;
    int l16  = lane & 15;
    int hi   = lane >> 4;
    int mbase = (blockIdx.x * 4 + wave) * 16;   // this wave's anchor rows

    // A fragments for the whole K=512 resident in 128 VGPRs
    long arow = (long)(mbase + l16);
    const __bf16* aptr = Pbf + arow * N_PROJ + hi * 8;
    v16bf af[16];
#pragma unroll
    for (int t = 0; t < 16; ++t) {
        FragBF u;
        u.h2[0] = *(const v8bf*)(aptr + t * 32);
        u.h2[1] = *(const v8bf*)(aptr + t * 32 + 16);
        af[t] = u.v;
    }

    float rz[8];
#pragma unroll
    for (int r = 0; r < 8; ++r)
        rz[r] = TEMP_INV / fmaxf(norms[mbase + hi * 8 + r], 1e-6f);

    float m[8], s[8], lin[8], dg[8];
#pragma unroll
    for (int r = 0; r < 8; ++r) { m[r] = -1e30f; s[r] = 0.f; lin[r] = 0.f; dg[r] = 0.f; }

    const float* nc = norms + B_ROWS;
    unsigned long long gbase =
        (unsigned long long)(unsigned long long)(const void*)(Pbf + (long)B_ROWS * N_PROJ);
    unsigned lds0 = (unsigned)(unsigned long long)(const void*)&bbuf[0][0][0];
    unsigned lds1 = (unsigned)(unsigned long long)(const void*)&bbuf[1][0][0];

    // prologue: stage tile 0
    if (wave == 0) {
        tdm_load_tile(lds0, gbase);
        __builtin_amdgcn_s_wait_tensorcnt(0);
    }
    __syncthreads();

    for (int ct = 0; ct < 512; ++ct) {
        int cur = ct & 1;
        // prefetch next wc tile into the other buffer (overlaps WMMA below)
        if (wave == 0 && ct + 1 < 512)
            tdm_load_tile(cur ? lds0 : lds1,
                          gbase + (unsigned long long)(ct + 1) * 16384ull);

        int jcol = ct * 16 + l16;
        float rc = 1.0f / fmaxf(nc[jcol], 1e-6f);
        const __bf16* bp = &bbuf[cur][l16][hi * 8];
        v8f c = {};
#pragma unroll
        for (int t = 0; t < 16; ++t) {
            FragBF b;
            b.h2[0] = *(const v8bf*)(bp + t * 32);       // ds_load_b128
            b.h2[1] = *(const v8bf*)(bp + t * 32 + 16);  // ds_load_b128
            c = __builtin_amdgcn_wmma_f32_16x16x32_bf16(false, af[t], false, b.v,
                                                        (short)0, c, false, false);
        }
#pragma unroll
        for (int r = 0; r < 8; ++r) {
            float v = c[r] * rz[r] * rc;
            lin[r] += v;
            float nm = fmaxf(m[r], v);
            s[r] = s[r] * __expf(m[r] - nm) + __expf(v - nm);
            m[r] = nm;
            if (mbase + hi * 8 + r == jcol) dg[r] = v;
        }

        if (wave == 0 && ct + 1 < 512)
            __builtin_amdgcn_s_wait_tensorcnt(0);
        __syncthreads();   // next buffer ready; current buffer free for reuse
    }

    // merge across the 16 lanes sharing each row (stays within half-wave)
    float lsum = 0.f, dsum = 0.f, lnsum = 0.f;
#pragma unroll
    for (int r = 0; r < 8; ++r) {
        float mm = m[r], sv = s[r], lv = lin[r], dv = dg[r];
#pragma unroll
        for (int msk = 8; msk >= 1; msk >>= 1) {
            float om = __shfl_xor(mm, msk, 32);
            float os = __shfl_xor(sv, msk, 32);
            float nm = fmaxf(mm, om);
            sv = sv * __expf(mm - nm) + os * __expf(om - nm);
            mm = nm;
            lv += __shfl_xor(lv, msk, 32);
            dv += __shfl_xor(dv, msk, 32);
        }
        lsum  += mm + __logf(sv) - dv;   // logsumexp - sim_ii
        dsum  += dv;
        lnsum += lv;
    }
    if (l16 == 0) {   // lanes 0 and 16: each holds 8 complete rows
        atomicAdd(acc + 0, lsum);
        atomicAdd(acc + 1, dsum);
        atomicAdd(acc + 2, lnsum);
    }
}

// ---------------------------------------------------------------- kernel 4
__global__ void k_final(const float* __restrict__ acc, float* __restrict__ out) {
    if (threadIdx.x == 0) {
        out[0] = acc[0] / 8192.0f;        // loss
        out[1] = acc[1] / 8192.0f;        // sim_pos
        out[2] = acc[2] / 67108864.0f;    // sim_mean (8192*8192)
    }
}

// ---------------------------------------------------------------- launch
extern "C" void kernel_launch(void* const* d_in, const int* in_sizes, int n_in,
                              void* d_out, int out_size, void* d_ws, size_t ws_size,
                              hipStream_t stream) {
    const float* h = (const float*)d_in[0];   // [16384, 2048]
    const float* W = (const float*)d_in[1];   // [512, 2048]
    float* out = (float*)d_out;               // 3 floats

    char* ws = (char*)d_ws;
    __bf16* Wbf  = (__bf16*)ws;                                  // 2,097,152 B
    __bf16* Pbf  = (__bf16*)(ws + 2097152);                      // 16,777,216 B
    float* norms = (float*)(ws + 2097152 + 16777216);            // 65,536 B
    float* acc   = (float*)(ws + 2097152 + 16777216 + 65536);    // 32 B

    k_convert_W<<<1024, 256, 0, stream>>>(W, Wbf);
    k_zero<<<1, 64, 0, stream>>>(acc);
    k_proj<<<4096, 256, 0, stream>>>(h, Wbf, Pbf);   // 32768 tiles, 8 waves/block
    k_norms<<<2048, 256, 0, stream>>>(Pbf, norms);   // 16384 rows
    k_sim<<<128, 128, 0, stream>>>(Pbf, norms, acc); // 512 anchor blocks, TDM-staged B
    k_final<<<1, 64, 0, stream>>>(acc, out);
}
